// WindowAttentionModule_17927193493643
// MI455X (gfx1250) — matrix operand order repaired and checked by reading, not compiled, not run on hardware
//
#include <hip/hip_runtime.h>

typedef __attribute__((ext_vector_type(16))) _Float16 v16h;
typedef __attribute__((ext_vector_type(8)))  float    v8f;

#define DIMC 512
#define NTOK 64
#define NHEADS 8
#define HD 64
#define TPB 256

union FragU {
    float4       f4[2];
    unsigned int u[8];
    v16h         h;
};

// ---- WMMA wrapper: D = A(16x32 f16) * B(32x16 f16) + C(16x16 f32) ----
__device__ inline v8f wmma_f16(v16h a, v16h b, v8f c) {
    return __builtin_amdgcn_wmma_f32_16x16x32_f16(
        /*neg_a=*/false, a, /*neg_b=*/false, b,
        /*c_mod=*/(short)0, c, /*reuse_a=*/false, /*reuse_b=*/false);
}

// A fragment (16x32) from row-major f16 [row][k], contiguous K order:
// lanes 0-15: row M=l, K k0..k0+15
// lanes16-31: row M=l-16, K k0+16..k0+31
__device__ inline v16h load_a_lds(const _Float16* base, int n0, int k0, int stride) {
    int l   = threadIdx.x & 31;
    int row = n0 + (l & 15);
    int kb  = k0 + ((l >> 4) << 4);
    const _Float16* p = base + row * stride + kb;
    FragU u;
    u.f4[0] = *(const float4*)(p);
    u.f4[1] = *(const float4*)(p + 8);
    return u.h;
}

// B fragment (32x16) from row-major f16 weights W[o][c] (B[c][o] = W[o][c]):
// lane holds column o = o0 + l%16; K = k0 + (l<16 ? 0..15 : 16..31), contiguous in memory.
__device__ inline v16h load_b_glb(const _Float16* w, int o0, int k0, int stride) {
    int l  = threadIdx.x & 31;
    int o  = o0 + (l & 15);
    int kb = k0 + ((l >> 4) << 4);
    const _Float16* p = w + o * stride + kb;
    FragU u;
    u.f4[0] = *(const float4*)(p);
    u.f4[1] = *(const float4*)(p + 8);
    return u.h;
}

// B fragment from pair-packed LDS: pk[c2][col] dword = (elem(2*c2,col), elem(2*c2+1,col)).
// VGPR j = K pair (base+2j, base+2j+1), base = k0 + (l<16 ? 0 : 16).
__device__ inline v16h load_b_pk(const unsigned int* pk, int c2base, int col0, int W) {
    int l   = threadIdx.x & 31;
    int col = col0 + (l & 15);
    int c2  = c2base + ((l >> 4) << 3);
    FragU u;
#pragma unroll
    for (int j = 0; j < 8; ++j) u.u[j] = pk[(c2 + j) * W + col];
    return u.h;
}

// Store D (f32 16x16) to f16 LDS row-major [row][col] with bias+scale.
__device__ inline void store_d_lds_h(_Float16* dst, int n0, int o0, int stride,
                                     v8f acc, float bias, float scale) {
    int l   = threadIdx.x & 31;
    int col = o0 + (l & 15);
    int rb  = n0 + ((l >> 4) << 3);
#pragma unroll
    for (int r = 0; r < 8; ++r)
        dst[(rb + r) * stride + col] = (_Float16)((acc[r] + bias) * scale);
}

// ---- prepass: f32 -> f16 weight conversion (pw_q, pw_k, pw_v, proj_w) into d_ws ----
__global__ __launch_bounds__(TPB)
void cvt_weights_kernel(const float* __restrict__ pwq, const float* __restrict__ pwk,
                        const float* __restrict__ pwv, const float* __restrict__ pjw,
                        _Float16* __restrict__ w16) {
    const float* srcs[4] = {pwq, pwk, pwv, pjw};
    int i   = (blockIdx.x * TPB + threadIdx.x) * 4;   // 1024*256*4 = 4 * 512*512
    int s   = i >> 18;
    int off = i & ((1 << 18) - 1);
    float4 v4 = *(const float4*)(srcs[s] + off);
    _Float16* d = w16 + i;
    d[0] = (_Float16)v4.x; d[1] = (_Float16)v4.y;
    d[2] = (_Float16)v4.z; d[3] = (_Float16)v4.w;
}

// ---- main kernel: one block = one window ----
__global__ __launch_bounds__(TPB)
void win_attn_kernel(const float* __restrict__ x,
                     const float* __restrict__ dw_q, const float* __restrict__ dwb_q,
                     const float* __restrict__ pwb_q,
                     const float* __restrict__ dw_k, const float* __restrict__ dwb_k,
                     const float* __restrict__ pwb_k,
                     const float* __restrict__ dw_v, const float* __restrict__ dwb_v,
                     const float* __restrict__ pwb_v,
                     const _Float16* __restrict__ w16,     // [4][512*512] f16: q,k,v,proj
                     const float* __restrict__ rpb,        // (225, 8)
                     const int*   __restrict__ relidx,     // (64, 64)
                     const float* __restrict__ projb,
                     float* __restrict__ out) {
    extern __shared__ char sm[];
    // 5 x 64KB regions, phase-reused:
    _Float16* xs   = (_Float16*)(sm);              // A: x f16      -> later kp / O
    _Float16* tb   = (_Float16*)(sm + 65536);      // B: dwconv out -> later vp
    _Float16* qb   = (_Float16*)(sm + 131072);     // C: q f16      -> later P f16
    _Float16* kbuf = (_Float16*)(sm + 196608);     // D: k f16      -> later scores (f32 lo)
    _Float16* vbuf = (_Float16*)(sm + 262144);     // E: v f16      -> later scores (f32 hi)
    float*    sbuf = (float*)(sm + 196608);        // D+E: 8*64*64 f32 scores

    const int tid  = threadIdx.x;
    const int wave = tid >> 5;
    const int rs   = wave & 3;       // row strip for 512-wide GEMMs
    const int cg   = wave >> 2;      // col half (0/1) for 512-wide GEMMs
    const int b    = blockIdx.x;
    const float* xb = x + (size_t)b * NTOK * DIMC;

    // ---- Phase 1: load x (f32 -> f16 [n][c]) ----
    for (int i = tid; i < NTOK * DIMC / 4; i += TPB) {
        float4 v4 = *(const float4*)(xb + i * 4);
        _Float16* p = xs + i * 4;
        p[0] = (_Float16)v4.x; p[1] = (_Float16)v4.y;
        p[2] = (_Float16)v4.z; p[3] = (_Float16)v4.w;
    }
    __syncthreads();

    // ---- Phase 2: q/k/v = pointwise(relu(depthwise(x))) ----
    const float* dws[3]  = {dw_q, dw_k, dw_v};
    const float* dwbs[3] = {dwb_q, dwb_k, dwb_v};
    const float* pwbs[3] = {pwb_q, pwb_k, pwb_v};
    _Float16*    dsts[3] = {qb, kbuf, vbuf};

    for (int t = 0; t < 3; ++t) {
        // depthwise 3x3 (pad 1) + bias + relu : xs -> tb
        const float* dw  = dws[t];
        const float* dwb = dwbs[t];
#pragma unroll
        for (int cc = 0; cc < 2; ++cc) {
            int c = tid + cc * TPB;
            float w9[9];
#pragma unroll
            for (int j = 0; j < 9; ++j) w9[j] = dw[c * 9 + j];
            float bias = dwb[c];
            for (int py = 0; py < 8; ++py) {
                for (int px = 0; px < 8; ++px) {
                    float acc = bias;
#pragma unroll
                    for (int dy = 0; dy < 3; ++dy) {
                        int yy = py + dy - 1;
                        if (yy < 0 || yy > 7) continue;
#pragma unroll
                        for (int dx = 0; dx < 3; ++dx) {
                            int xx = px + dx - 1;
                            if (xx < 0 || xx > 7) continue;
                            acc += w9[dy * 3 + dx] * (float)xs[(yy * 8 + xx) * DIMC + c];
                        }
                    }
                    tb[(py * 8 + px) * DIMC + c] = (_Float16)fmaxf(acc, 0.f);
                }
            }
        }
        __syncthreads();

        // pointwise GEMM: dst[64][512] = tb[64][512] x W^T, 4-wide register blocking:
        // each wave owns row strip rs, col tiles cg*16 + g*4 .. +3 -> A frag shared by 4 WMMAs.
        const _Float16* wB  = w16 + (size_t)t * DIMC * DIMC;
        const float*    pwb = pwbs[t];
        const float     scl = (t == 0) ? 0.125f : 1.0f;   // q pre-scaled by HEAD_DIM^-0.5
        _Float16*       dst = dsts[t];
        int n0 = rs << 4;
        for (int g = 0; g < 4; ++g) {
            int o0 = (cg * 16 + g * 4) << 4;     // base of 4 consecutive 16-col tiles
            v8f a0 = {}, a1 = {}, a2 = {}, a3 = {};
            for (int k0 = 0; k0 < DIMC; k0 += 32) {
                if (k0 + 32 < DIMC)
                    __builtin_prefetch((const void*)(wB + (size_t)(o0 + (tid & 15)) * DIMC + k0 + 32), 0, 1);
                v16h a = load_a_lds(tb, n0, k0, DIMC);
                a0 = wmma_f16(a, load_b_glb(wB, o0,      k0, DIMC), a0);
                a1 = wmma_f16(a, load_b_glb(wB, o0 + 16, k0, DIMC), a1);
                a2 = wmma_f16(a, load_b_glb(wB, o0 + 32, k0, DIMC), a2);
                a3 = wmma_f16(a, load_b_glb(wB, o0 + 48, k0, DIMC), a3);
            }
            store_d_lds_h(dst, n0, o0,      DIMC, a0, pwb[o0      + (tid & 15)], scl);
            store_d_lds_h(dst, n0, o0 + 16, DIMC, a1, pwb[o0 + 16 + (tid & 15)], scl);
            store_d_lds_h(dst, n0, o0 + 32, DIMC, a2, pwb[o0 + 32 + (tid & 15)], scl);
            store_d_lds_h(dst, n0, o0 + 48, DIMC, a3, pwb[o0 + 48 + (tid & 15)], scl);
        }
        __syncthreads();
    }

    // ---- Phase 3: repack k -> kp[c/2][m], v -> vp[m/2][c] (pair-packed dwords) ----
    unsigned int* kp = (unsigned int*)sm;            // region A (xs dead)
    unsigned int* vp = (unsigned int*)(sm + 65536);  // region B (tb dead)
    {
        const unsigned int* ksrc = (const unsigned int*)kbuf;  // [m][256] dwords
        for (int i = tid; i < 256 * 64; i += TPB) {
            int c2 = i >> 6, m = i & 63;
            kp[i] = ksrc[m * 256 + c2];              // (k[m][2c2], k[m][2c2+1])
        }
        const unsigned int* vsrc = (const unsigned int*)vbuf;  // [m][256] dwords
        for (int i = tid; i < 32 * 256; i += TPB) {
            int m2 = i >> 8, cp = i & 255;
            unsigned int dA = vsrc[(2 * m2) * 256 + cp];
            unsigned int dB = vsrc[(2 * m2 + 1) * 256 + cp];
            vp[m2 * 512 + 2 * cp]     = (dA & 0xffffu) | (dB << 16);
            vp[m2 * 512 + 2 * cp + 1] = (dA >> 16) | (dB & 0xffff0000u);
        }
    }
    __syncthreads();

    // ---- Phase 4: S = (q*scale) . k^T + relpos bias. One head per wave. ----
    {
        const int h = wave;
        for (int trs = 0; trs < 4; ++trs) {
            int n0 = trs << 4;
            v8f a0 = {}, a1 = {}, a2 = {}, a3 = {};
#pragma unroll
            for (int d0 = 0; d0 < HD; d0 += 32) {
                v16h a = load_a_lds(qb, n0, h * HD + d0, DIMC);
                int c2 = (h * HD + d0) >> 1;
                a0 = wmma_f16(a, load_b_pk(kp, c2, 0,  64), a0);
                a1 = wmma_f16(a, load_b_pk(kp, c2, 16, 64), a1);
                a2 = wmma_f16(a, load_b_pk(kp, c2, 32, 64), a2);
                a3 = wmma_f16(a, load_b_pk(kp, c2, 48, 64), a3);
            }
            int l  = tid & 31;
            int rb = n0 + ((l >> 4) << 3);
            v8f accs[4] = {a0, a1, a2, a3};
#pragma unroll
            for (int cs = 0; cs < 4; ++cs) {
                int m = (cs << 4) + (l & 15);
#pragma unroll
                for (int r = 0; r < 8; ++r) {
                    int n = rb + r;
                    float bias = rpb[relidx[n * 64 + m] * NHEADS + h];
                    sbuf[(h * 64 + n) * 64 + m] = accs[cs][r] + bias;
                }
            }
        }
    }
    __syncthreads();

    // ---- Phase 5: row softmax over m; P f16 -> region C (q dead) ----
    _Float16* pb = qb;   // [h][n][m], stride 64
#pragma unroll
    for (int rr = 0; rr < 2; ++rr) {
        int row = tid + rr * TPB;                   // 512 rows = h*64+n
        float* p = sbuf + row * 64;
        float mx = -1e30f;
        for (int m = 0; m < 64; ++m) mx = fmaxf(mx, p[m]);
        float s = 0.f;
        for (int m = 0; m < 64; ++m) s += __expf(p[m] - mx);
        float inv = 1.f / s;
        _Float16* prow = pb + row * 64;
        for (int m = 0; m < 64; ++m) prow[m] = (_Float16)(__expf(p[m] - mx) * inv);
    }
    __syncthreads();

    // ---- Phase 6: O = P . V, f16 [n][c] -> region A (kp dead). One head per wave. ----
    _Float16* ob = (_Float16*)sm;
    {
        const int h = wave;
        for (int trs = 0; trs < 4; ++trs) {
            int n0 = trs << 4;
            v8f a0 = {}, a1 = {}, a2 = {}, a3 = {};
#pragma unroll
            for (int m0 = 0; m0 < 64; m0 += 32) {
                v16h a = load_a_lds(pb + h * 64 * 64, n0, m0, 64);
                int m2 = m0 >> 1;
                a0 = wmma_f16(a, load_b_pk(vp, m2, h * HD,      512), a0);
                a1 = wmma_f16(a, load_b_pk(vp, m2, h * HD + 16, 512), a1);
                a2 = wmma_f16(a, load_b_pk(vp, m2, h * HD + 32, 512), a2);
                a3 = wmma_f16(a, load_b_pk(vp, m2, h * HD + 48, 512), a3);
            }
            store_d_lds_h(ob, n0, h * HD,      DIMC, a0, 0.f, 1.f);
            store_d_lds_h(ob, n0, h * HD + 16, DIMC, a1, 0.f, 1.f);
            store_d_lds_h(ob, n0, h * HD + 32, DIMC, a2, 0.f, 1.f);
            store_d_lds_h(ob, n0, h * HD + 48, DIMC, a3, 0.f, 1.f);
        }
    }
    __syncthreads();

    // ---- Phase 7: out = O . proj_w^T + proj_b (f32 to global) ----
    const _Float16* wP   = w16 + (size_t)3 * DIMC * DIMC;
    float*          outb = out + (size_t)b * NTOK * DIMC;
    {
        int n0 = rs << 4;
        for (int g = 0; g < 4; ++g) {
            int o0 = (cg * 16 + g * 4) << 4;
            v8f a0 = {}, a1 = {}, a2 = {}, a3 = {};
            for (int k0 = 0; k0 < DIMC; k0 += 32) {
                if (k0 + 32 < DIMC)
                    __builtin_prefetch((const void*)(wP + (size_t)(o0 + (tid & 15)) * DIMC + k0 + 32), 0, 1);
                v16h a = load_a_lds(ob, n0, k0, DIMC);
                a0 = wmma_f16(a, load_b_glb(wP, o0,      k0, DIMC), a0);
                a1 = wmma_f16(a, load_b_glb(wP, o0 + 16, k0, DIMC), a1);
                a2 = wmma_f16(a, load_b_glb(wP, o0 + 32, k0, DIMC), a2);
                a3 = wmma_f16(a, load_b_glb(wP, o0 + 48, k0, DIMC), a3);
            }
            int l  = tid & 31;
            int rb = n0 + ((l >> 4) << 3);
            v8f accs[4] = {a0, a1, a2, a3};
#pragma unroll
            for (int ct = 0; ct < 4; ++ct) {
                int col = o0 + (ct << 4) + (l & 15);
                float bias = projb[col];
#pragma unroll
                for (int r = 0; r < 8; ++r)
                    outb[(rb + r) * DIMC + col] = accs[ct][r] + bias;
            }
        }
    }
}

extern "C" void kernel_launch(void* const* d_in, const int* in_sizes, int n_in,
                              void* d_out, int out_size, void* d_ws, size_t ws_size,
                              hipStream_t stream) {
    const float* x      = (const float*)d_in[0];
    const float* dw_q   = (const float*)d_in[1];
    const float* dwb_q  = (const float*)d_in[2];
    const float* pw_q   = (const float*)d_in[3];
    const float* pwb_q  = (const float*)d_in[4];
    const float* dw_k   = (const float*)d_in[5];
    const float* dwb_k  = (const float*)d_in[6];
    const float* pw_k   = (const float*)d_in[7];
    const float* pwb_k  = (const float*)d_in[8];
    const float* dw_v   = (const float*)d_in[9];
    const float* dwb_v  = (const float*)d_in[10];
    const float* pw_v   = (const float*)d_in[11];
    const float* pwb_v  = (const float*)d_in[12];
    const float* rpb    = (const float*)d_in[13];
    const int*   relidx = (const int*)d_in[14];
    const float* proj_w = (const float*)d_in[15];
    const float* proj_b = (const float*)d_in[16];
    float*       out    = (float*)d_out;

    _Float16* w16 = (_Float16*)d_ws;   // 4 * 512*512 f16 = 2 MB

    cvt_weights_kernel<<<1024, TPB, 0, stream>>>(pw_q, pw_k, pw_v, proj_w, w16);

    win_attn_kernel<<<1024, TPB, 320 * 1024, stream>>>(
        x, dw_q, dwb_q, pwb_q, dw_k, dwb_k, pwb_k, dw_v, dwb_v, pwb_v,
        w16, rpb, relidx, proj_b, out);
}
